// LanguageModel_58574763983800
// MI455X (gfx1250) — compile-verified, matrix operand order checked by reading, not run on hardware
//
#include <hip/hip_runtime.h>

// ---------------------------------------------------------------------------
// Types / helpers
// ---------------------------------------------------------------------------
typedef __attribute__((ext_vector_type(16))) __bf16 v16bf;
typedef __attribute__((ext_vector_type(8)))  float  v8f;
typedef unsigned short u16;

struct U128 { unsigned int x, y, z, w; };          // POD 16-byte packet

union Frag {                                        // bf16 WMMA operand (8 VGPRs)
    v16bf v;
    U128  q[2];
    u16   u[16];
};

union Pack8 {                                       // 8 bf16 in one 16B packet
    U128 q;
    u16  u[8];
};

__device__ __forceinline__ u16 f2bf(float f) {
    unsigned int u = __float_as_uint(f);
    unsigned int r = (u + 0x7FFFu + ((u >> 16) & 1u)) >> 16;   // RNE
    return (u16)r;
}

// Fragment: 16 bf16 of one LDS row; elements 0..7 = base+0..7,
// elements 8..15 = base+16..23 (CDNA5 16-bit A/B operand layout; the
// caller supplies base = row*stride + k_tile + 8*(lane>=16)).
__device__ __forceinline__ v16bf frag_row(const u16* p) {
    Frag f;
    f.q[0] = *(const U128*)(p);
    f.q[1] = *(const U128*)(p + 16);
    return f.v;
}

__device__ __forceinline__ v8f wmma_bf16(v16bf a, v16bf b, v8f c) {
    return __builtin_amdgcn_wmma_f32_16x16x32_bf16(
        /*neg_a=*/false, a, /*neg_b=*/false, b,
        /*c_mod=*/(short)0, c, /*reuse_a=*/false, /*reuse_b=*/false);
}

// CDNA5 async memory->LDS copy (16B per lane), tracked by ASYNCcnt.
// vdst = LDS byte offset (low 32 bits of a generic LDS pointer, ISA 10.2),
// vaddr = 64-bit global address, GV mode.
__device__ __forceinline__ void async_ld16(const u16* lds_p, const u16* gp) {
    unsigned lds = (unsigned)(size_t)lds_p;
    unsigned long long ga = (unsigned long long)(size_t)gp;
    asm volatile("global_load_async_to_lds_b128 %0, %1, off"
                 :
                 : "v"(lds), "v"(ga)
                 : "memory");
}
__device__ __forceinline__ void wait_async_prev() {   // allow 12 in flight
    asm volatile("s_wait_asynccnt 12" ::: "memory");
}
__device__ __forceinline__ void wait_async_all() {
    asm volatile("s_wait_asynccnt 0" ::: "memory");
}

#define ZERO8 {0.f,0.f,0.f,0.f,0.f,0.f,0.f,0.f}

// ---------------------------------------------------------------------------
// Model constants
// ---------------------------------------------------------------------------
#define NB   128
#define NT   64
#define NC   384
#define NL   6
#define NH   4
#define NDH  96
#define NV   32000
#define NFF  1536
#define NM   (NB * NT)          /* 8192 rows */
#define HD   (NH * NDH)         /* 384 */

// ---------------------------------------------------------------------------
// Tiled transpose + f32->bf16 convert:  src (nmat x [R,Q] f32, row-major)
// -> dst (nmat x [Q,R] bf16, row-major).  Coalesced on both sides via LDS.
// R, Q multiples of 32.
// ---------------------------------------------------------------------------
__global__ __launch_bounds__(256) void k_convert_t(const float* __restrict__ src,
                                                   u16* __restrict__ dst,
                                                   int R, int Q) {
    __shared__ float t[32][33];
    long base = (long)blockIdx.z * R * Q;
    int rb = blockIdx.y * 32, qb = blockIdx.x * 32;
    int tx = threadIdx.x, ty = threadIdx.y;            // blockDim = (32, 8)
    for (int j = ty; j < 32; j += 8)
        t[j][tx] = src[base + (long)(rb + j) * Q + qb + tx];
    __syncthreads();
    for (int j = ty; j < 32; j += 8)
        dst[base + (long)(qb + j) * R + rb + tx] = f2bf(t[tx][j]);
}

// ---------------------------------------------------------------------------
// Embedding: x[bt,c] = tok_emb[idx[bt],c] + pos_emb[t,c]
// ---------------------------------------------------------------------------
__global__ void k_embed(const int* __restrict__ idx, const float* __restrict__ tok,
                        const float* __restrict__ pos, float* __restrict__ x, int total) {
    int i = blockIdx.x * blockDim.x + threadIdx.x;
    if (i >= total) return;
    int bt = i / NC, c = i % NC, t = bt % NT;
    x[i] = tok[(long)idx[bt] * NC + c] + pos[t * NC + c];
}

// ---------------------------------------------------------------------------
// LayerNorm over C=384, one block (128 threads) per row, bf16 output
// ---------------------------------------------------------------------------
__global__ __launch_bounds__(128) void k_layernorm(const float* __restrict__ x,
                                                   const float* __restrict__ g,
                                                   const float* __restrict__ b,
                                                   u16* __restrict__ out) {
    __shared__ float s1[128], s2[128];
    int row = blockIdx.x, tid = threadIdx.x;
    const float* xr = x + (long)row * NC;
    float v0 = xr[tid], v1 = xr[tid + 128], v2 = xr[tid + 256];
    s1[tid] = v0 + v1 + v2;
    s2[tid] = v0 * v0 + v1 * v1 + v2 * v2;
    __syncthreads();
    for (int off = 64; off > 0; off >>= 1) {
        if (tid < off) { s1[tid] += s1[tid + off]; s2[tid] += s2[tid + off]; }
        __syncthreads();
    }
    float mu  = s1[0] * (1.f / NC);
    float var = s2[0] * (1.f / NC) - mu * mu;
    float rs  = rsqrtf(var + 1e-5f);
    u16* o = out + (long)row * NC;
    o[tid]       = f2bf((v0 - mu) * rs * g[tid]       + b[tid]);
    o[tid + 128] = f2bf((v1 - mu) * rs * g[tid + 128] + b[tid + 128]);
    o[tid + 256] = f2bf((v2 - mu) * rs * g[tid + 256] + b[tid + 256]);
}

// ---------------------------------------------------------------------------
// bf16 WMMA GEMM: C[M,N] = A[M,K] @ B[K,N] (+bias)(+ReLU)(+residual)
// B is stored TRANSPOSED ([N,K] row-major) so both operands use the same
// contiguous fragment layout (2x ds_load_b128 each, no scalar gathers).
// Block = 128 threads (4 waves). Block tile 128x64, K-tile 64, DOUBLE-
// BUFFERED via GLOBAL_LOAD_ASYNC_TO_LDS_B128: the next tile's 12 async
// packets/thread stream into LDS while WMMAs consume the current one
// (s_wait_asynccnt 12 waits only for the previous tile; same-counter ops
// complete in order).  Each wave owns a 32x64 strip -> 1.5 DS loads/WMMA.
// M multiple of 128; N multiple of 64; K multiple of 64 (true here).
// ---------------------------------------------------------------------------
#define KT  64
#define LDT 72   /* padded LDS row stride (u16): 16 lanes -> 16 distinct banks */

__global__ __launch_bounds__(128) void k_gemm(const u16* __restrict__ A,
                                              const u16* __restrict__ Bt,
                                              const float* __restrict__ bias,
                                              const float* __restrict__ resid,
                                              float* __restrict__ outF,
                                              u16* __restrict__ outB,
                                              int M, int N, int K, int relu) {
    __shared__ __align__(16) u16 As[2][128 * LDT];
    __shared__ __align__(16) u16 Bs[2][64 * LDT];
    int n0 = blockIdx.x * 64, m0 = blockIdx.y * 128;
    int tid = threadIdx.x, wave = tid >> 5, lane = tid & 31;
    int l15 = lane & 15, hi8 = (lane >> 4) << 3;

    v8f acc[2][4] = {{ZERO8, ZERO8, ZERO8, ZERO8},
                     {ZERO8, ZERO8, ZERO8, ZERO8}};

    // issue one K-tile's async loads (12 x 16B packets per thread)
    auto issue = [&](int bi, int kk) {
#pragma unroll
        for (int i = 0; i < 8; ++i) {                // A: 128 rows x 64 K
            int idx = tid + i * 128;
            int r = idx >> 3, c8 = (idx & 7) * 8;
            async_ld16(&As[bi][r * LDT + c8], A + (long)(m0 + r) * K + kk + c8);
        }
#pragma unroll
        for (int i = 0; i < 4; ++i) {                // B^T: 64 rows x 64 K
            int idx = tid + i * 128;
            int r = idx >> 3, c8 = (idx & 7) * 8;
            async_ld16(&Bs[bi][r * LDT + c8], Bt + (long)(n0 + r) * K + kk + c8);
        }
    };

    int ntile = K / KT;
    issue(0, 0);
    for (int t = 0; t < ntile; ++t) {
        int bi = t & 1;
        if (t + 1 < ntile) {
            issue(bi ^ 1, (t + 1) * KT);             // prefetch next tile
            wait_async_prev();                       // tile t's packets landed
        } else {
            wait_async_all();
        }
        __syncthreads();
#pragma unroll
        for (int k2 = 0; k2 < KT; k2 += 32) {
            int kb = k2 + hi8;
            v16bf a0 = frag_row(&As[bi][(wave * 32      + l15) * LDT + kb]);
            v16bf a1 = frag_row(&As[bi][(wave * 32 + 16 + l15) * LDT + kb]);
#pragma unroll
            for (int s = 0; s < 4; ++s) {
                v16bf bf = frag_row(&Bs[bi][(s * 16 + l15) * LDT + kb]);
                acc[0][s] = wmma_bf16(a0, bf, acc[0][s]);
                acc[1][s] = wmma_bf16(a1, bf, acc[1][s]);
            }
        }
        __syncthreads();
    }

#pragma unroll
    for (int t = 0; t < 2; ++t) {
#pragma unroll
        for (int s = 0; s < 4; ++s) {
#pragma unroll
            for (int r = 0; r < 8; ++r) {
                int m = m0 + wave * 32 + t * 16 + r + hi8;   // C/D: VGPR r, +8 upper lanes
                int n = n0 + s * 16 + l15;
                float v = acc[t][s][r];
                if (bias)  v += bias[n];
                if (relu)  v = fmaxf(v, 0.f);
                if (resid) v += resid[(long)m * N + n];
                if (outF)  outF[(long)m * N + n] = v;
                if (outB)  outB[(long)m * N + n] = f2bf(v);
            }
        }
    }
}

// ---------------------------------------------------------------------------
// Fused causal attention: one block per (b,h). q,k staged row-major, v staged
// TRANSPOSED (vt[d][s]) so the P@V B-fragments are contiguous too.
// S = q k^T (WMMA), causal softmax, att = P v (WMMA), bf16 head-concat out.
// P reuses the q tile (q dead after the score phase) to stay < 64KB LDS.
// ---------------------------------------------------------------------------
#define QLD 104  /* 96 + 8 pad  */
#define VLD 72   /* 64 + 8 pad  */
#define SLD 65   /* 64 + 1 pad (f32) */

__global__ __launch_bounds__(128) void k_attention(const u16* __restrict__ Q,
                                                   const u16* __restrict__ Kb,
                                                   const u16* __restrict__ Vb,
                                                   u16* __restrict__ att,
                                                   float scale) {
    __shared__ __align__(16) u16 qs[64 * QLD];      // q tile; reused as P tile
    __shared__ __align__(16) u16 ks[64 * QLD];
    __shared__ __align__(16) u16 vt[96 * VLD];      // v transposed: vt[d][s]
    __shared__ float Sf[64 * SLD];

    int bh = blockIdx.x;
    int b = bh >> 2, h = bh & 3;
    int tid = threadIdx.x, wave = tid >> 5, lane = tid & 31;
    int l15 = lane & 15, hi8 = (lane >> 4) << 3;
    long base = (long)b * NT * NC + h * NDH;

    // stage q,k (row-major) and v (transposed) tiles [64 x 96]
    for (int i = tid; i < 64 * 12; i += 128) {
        int r = i / 12, c8 = (i % 12) * 8;
        long g = base + (long)r * NC + c8;
        *(U128*)(qs + r * QLD + c8) = *(const U128*)(Q  + g);
        *(U128*)(ks + r * QLD + c8) = *(const U128*)(Kb + g);
        Pack8 pv; pv.q = *(const U128*)(Vb + g);
#pragma unroll
        for (int j = 0; j < 8; ++j) vt[(c8 + j) * VLD + r] = pv.u[j];
    }
    __syncthreads();

    // S = q k^T * scale ; wave w -> rows [w*16, w*16+16)
    int mrow = wave * 16;
    {
        v8f acc[4] = {ZERO8, ZERO8, ZERO8, ZERO8};
        for (int kk = 0; kk < 96; kk += 32) {
            int kb = kk + hi8;
            v16bf a = frag_row(qs + (mrow + l15) * QLD + kb);
#pragma unroll
            for (int s = 0; s < 4; ++s) {
                // B column n = k-row n : same contiguous layout as A
                v16bf bf = frag_row(ks + (s * 16 + l15) * QLD + kb);
                acc[s] = wmma_bf16(a, bf, acc[s]);
            }
        }
#pragma unroll
        for (int s = 0; s < 4; ++s)
#pragma unroll
            for (int r = 0; r < 8; ++r)
                Sf[(mrow + r + hi8) * SLD + s * 16 + l15] = acc[s][r] * scale;
    }
    __syncthreads();

    // causal softmax per row (threads 0..63); write P into the q tile
    if (tid < 64) {
        int t = tid;
        float mx = -1e30f;
        for (int s = 0; s <= t; ++s) mx = fmaxf(mx, Sf[t * SLD + s]);
        float sum = 0.f;
        for (int s = 0; s <= t; ++s) {
            float e = __expf(Sf[t * SLD + s] - mx);
            Sf[t * SLD + s] = e;
            sum += e;
        }
        float inv = 1.f / sum;
        for (int s = 0; s < 64; ++s)
            qs[t * QLD + s] = (s <= t) ? f2bf(Sf[t * SLD + s] * inv) : (u16)0;
    }
    __syncthreads();

    // att = P @ v  [64 x 96]
    {
        v8f acc[6] = {ZERO8, ZERO8, ZERO8, ZERO8, ZERO8, ZERO8};
        for (int kk = 0; kk < 64; kk += 32) {
            int kb = kk + hi8;
            v16bf a = frag_row(qs + (mrow + l15) * QLD + kb);   // P fragment
#pragma unroll
            for (int n = 0; n < 6; ++n) {
                v16bf bf = frag_row(vt + (n * 16 + l15) * VLD + kb);
                acc[n] = wmma_bf16(a, bf, acc[n]);
            }
        }
#pragma unroll
        for (int n = 0; n < 6; ++n)
#pragma unroll
            for (int r = 0; r < 8; ++r) {
                int m = mrow + r + hi8;
                att[base + (long)m * NC + n * 16 + l15] = f2bf(acc[n][r]);
            }
    }
}

// ---------------------------------------------------------------------------
// Loss: per row log-softmax over V=32000, gather target, mean-NLL via atomics
// ---------------------------------------------------------------------------
__global__ void k_zero(float* p) { p[0] = 0.f; }

__global__ __launch_bounds__(256) void k_loss(const float* __restrict__ logits,
                                              const int* __restrict__ targets,
                                              float* __restrict__ loss) {
    __shared__ float sh[256];
    int row = blockIdx.x, tid = threadIdx.x;
    const float* lr = logits + (long)row * NV;
    float mx = -1e30f;
    for (int i = tid; i < NV; i += 256) mx = fmaxf(mx, lr[i]);
    sh[tid] = mx; __syncthreads();
    for (int off = 128; off > 0; off >>= 1) {
        if (tid < off) sh[tid] = fmaxf(sh[tid], sh[tid + off]);
        __syncthreads();
    }
    mx = sh[0]; __syncthreads();
    float sum = 0.f;
    for (int i = tid; i < NV; i += 256) sum += __expf(lr[i] - mx);
    sh[tid] = sum; __syncthreads();
    for (int off = 128; off > 0; off >>= 1) {
        if (tid < off) sh[tid] += sh[tid + off];
        __syncthreads();
    }
    if (tid == 0) {
        float logZ = mx + __logf(sh[0]);
        float lp = lr[targets[row]] - logZ;
        atomicAdd(loss, -lp * (1.0f / (float)NM));
    }
}

// ---------------------------------------------------------------------------
// Host driver
// ---------------------------------------------------------------------------
extern "C" void kernel_launch(void* const* d_in, const int* in_sizes, int n_in,
                              void* d_out, int out_size, void* d_ws, size_t ws_size,
                              hipStream_t stream) {
    const int*   idx     = (const int*)  d_in[0];
    const int*   targets = (const int*)  d_in[1];
    const float* tok_emb = (const float*)d_in[2];
    const float* pos_emb = (const float*)d_in[3];
    const float* wq      = (const float*)d_in[4];
    const float* wk      = (const float*)d_in[5];
    const float* wv      = (const float*)d_in[6];
    const float* w_proj  = (const float*)d_in[7];
    const float* b_proj  = (const float*)d_in[8];
    const float* ln1_g   = (const float*)d_in[9];
    const float* ln1_b   = (const float*)d_in[10];
    const float* ln2_g   = (const float*)d_in[11];
    const float* ln2_b   = (const float*)d_in[12];
    const float* w1      = (const float*)d_in[13];
    const float* b1      = (const float*)d_in[14];
    const float* w2      = (const float*)d_in[15];
    const float* b2      = (const float*)d_in[16];
    const float* lnf_g   = (const float*)d_in[17];
    const float* lnf_b   = (const float*)d_in[18];
    const float* w_lm    = (const float*)d_in[19];
    const float* b_lm    = (const float*)d_in[20];

    float* out   = (float*)d_out;
    float* lossp = out + (size_t)NM * NV;

    // workspace carve-out (256B aligned)
    char* ws = (char*)d_ws;
    auto alloc = [&](size_t bytes) -> char* {
        char* p = ws;
        ws += (bytes + 255) & ~(size_t)255;
        return p;
    };
    float* x    = (float*)alloc((size_t)NM * NC * 4);       // residual stream f32
    u16*   hbuf = (u16*)  alloc((size_t)NM * NC * 2);       // LN output (bf16)
    u16*   qb   = (u16*)  alloc((size_t)NM * NC * 2);
    u16*   kb   = (u16*)  alloc((size_t)NM * NC * 2);
    u16*   vb   = (u16*)  alloc((size_t)NM * NC * 2);
    u16*   attb = (u16*)  alloc((size_t)NM * NC * 2);
    u16*   ffb  = (u16*)  alloc((size_t)NM * NFF * 2);
    // all weights stored TRANSPOSED ([N,K] bf16)
    u16*   Wq   = (u16*)  alloc((size_t)NL * HD * NC * 2);
    u16*   Wk   = (u16*)  alloc((size_t)NL * HD * NC * 2);
    u16*   Wv   = (u16*)  alloc((size_t)NL * HD * NC * 2);
    u16*   Wp   = (u16*)  alloc((size_t)NL * NC * NC * 2);
    u16*   W1   = (u16*)  alloc((size_t)NL * NFF * NC * 2);
    u16*   W2   = (u16*)  alloc((size_t)NL * NC * NFF * 2);
    u16*   Wlm  = (u16*)  alloc((size_t)NV * NC * 2);

    const float SCALE = (float)NDH / sqrtf((float)NC);
    dim3 cblk(32, 8);

    // ---- weight conversion (transpose + bf16) ----
    // wq/wk/wv: per (l,h) transpose [C,Dh] -> [Dh,C]; concatenation over h
    // yields [L, H*Dh, C], exactly B^T for the fused QKV GEMMs.
    {
        dim3 gqkv(NDH / 32, NC / 32, NL * NH);
        k_convert_t<<<gqkv, cblk, 0, stream>>>(wq, Wq, NC, NDH);
        k_convert_t<<<gqkv, cblk, 0, stream>>>(wk, Wk, NC, NDH);
        k_convert_t<<<gqkv, cblk, 0, stream>>>(wv, Wv, NC, NDH);
        dim3 gp(NC / 32, NC / 32, NL);
        k_convert_t<<<gp, cblk, 0, stream>>>(w_proj, Wp, NC, NC);
        dim3 g1(NFF / 32, NC / 32, NL);
        k_convert_t<<<g1, cblk, 0, stream>>>(w1, W1, NC, NFF);
        dim3 g2(NC / 32, NFF / 32, NL);
        k_convert_t<<<g2, cblk, 0, stream>>>(w2, W2, NFF, NC);
        dim3 gl(NV / 32, NC / 32, 1);
        k_convert_t<<<gl, cblk, 0, stream>>>(w_lm, Wlm, NC, NV);
    }

    // ---- embedding ----
    {
        int total = NM * NC;
        k_embed<<<(total + 255) / 256, 256, 0, stream>>>(idx, tok_emb, pos_emb, x, total);
    }

    dim3 blk(128);
    // ---- transformer layers ----
    for (int l = 0; l < NL; ++l) {
        k_layernorm<<<NM, blk, 0, stream>>>(x, ln1_g + l * NC, ln1_b + l * NC, hbuf);

        dim3 gq(HD / 64, NM / 128);
        k_gemm<<<gq, blk, 0, stream>>>(hbuf, Wq + (size_t)l * HD * NC, nullptr, nullptr,
                                       nullptr, qb, NM, HD, NC, 0);
        k_gemm<<<gq, blk, 0, stream>>>(hbuf, Wk + (size_t)l * HD * NC, nullptr, nullptr,
                                       nullptr, kb, NM, HD, NC, 0);
        k_gemm<<<gq, blk, 0, stream>>>(hbuf, Wv + (size_t)l * HD * NC, nullptr, nullptr,
                                       nullptr, vb, NM, HD, NC, 0);

        k_attention<<<NB * NH, blk, 0, stream>>>(qb, kb, vb, attb, SCALE);

        dim3 gp(NC / 64, NM / 128);
        k_gemm<<<gp, blk, 0, stream>>>(attb, Wp + (size_t)l * NC * NC, b_proj + l * NC, x,
                                       x, nullptr, NM, NC, NC, 0);

        k_layernorm<<<NM, blk, 0, stream>>>(x, ln2_g + l * NC, ln2_b + l * NC, hbuf);

        dim3 g1(NFF / 64, NM / 128);
        k_gemm<<<g1, blk, 0, stream>>>(hbuf, W1 + (size_t)l * NFF * NC, b1 + l * NFF, nullptr,
                                       nullptr, ffb, NM, NFF, NC, 1);

        dim3 g2(NC / 64, NM / 128);
        k_gemm<<<g2, blk, 0, stream>>>(ffb, W2 + (size_t)l * NC * NFF, b2 + l * NC, x,
                                       x, nullptr, NM, NC, NFF, 0);
    }

    // ---- final LN + LM head -> logits (d_out) ----
    k_layernorm<<<NM, blk, 0, stream>>>(x, lnf_g, lnf_b, hbuf);
    dim3 glm(NV / 64, NM / 128);
    k_gemm<<<glm, blk, 0, stream>>>(hbuf, Wlm, b_lm, nullptr, out, nullptr, NM, NV, NC, 0);

    // ---- loss ----
    k_zero<<<1, 1, 0, stream>>>(lossp);
    k_loss<<<NM, 256, 0, stream>>>(out, targets, lossp);
}
